// Reservoir_81200651698782
// MI455X (gfx1250) — matrix operand order repaired and checked by reading, not compiled, not run on hardware
//
#include <hip/hip_runtime.h>

typedef __attribute__((ext_vector_type(16))) _Float16 v16h;
typedef __attribute__((ext_vector_type(8)))  _Float16 v8h;
typedef __attribute__((ext_vector_type(8)))  float    v8f;
typedef __attribute__((ext_vector_type(4)))  float    v4f;

#define BB 8
#define SS 200
#define NN 100
#define FF 32
#define HH 128
#define BN 800          // B*N rows
#define NTILES 8        // H/16
#define ALPHA 0.9f

// f16 weight layout inside d_ws (offsets in halves)
#define OFF_IH0 0        // 128*32
#define OFF_HH0 4096     // 128*128
#define OFF_IH1 20480    // 128*128
#define OFF_HH1 36864    // 128*128

#if defined(__has_builtin) && __has_builtin(__builtin_amdgcn_tanhf)
__device__ __forceinline__ float fast_tanh(float x) { return __builtin_amdgcn_tanhf(x); }
#else
// branch-free fallback: v_exp_f32 based, no libm scalarization
__device__ __forceinline__ float fast_tanh(float x) {
    float e = __expf(2.f * x);
    return (e - 1.f) / (e + 1.f);
}
#endif

__global__ void cvt_f32_f16_kernel(const float* __restrict__ src,
                                   _Float16* __restrict__ dst, int n) {
    int i = blockIdx.x * blockDim.x + threadIdx.x;
    if (i < n) dst[i] = (_Float16)src[i];
}

// Load a 16x16 B-matrix fragment: f16 per lane (row = N, k = g*8..+7 and 16+g*8..+7)
__device__ __forceinline__ v16h load_frag16(const _Float16* p) {
    v8h lo = *(const v8h*)p;
    v8h hi = *(const v8h*)(p + 16);
    v16h r;
#pragma unroll
    for (int i = 0; i < 8; ++i) { r[i] = lo[i]; r[i + 8] = hi[i]; }
    return r;
}

// Load an A-matrix k-block fragment (16-bit A 16x32 layout) from f16 row-major [16][HH] LDS
__device__ __forceinline__ v16h load_a_frag(const _Float16* hrow_base, int n16, int g, int kk) {
    const _Float16* base = hrow_base + n16 * HH + kk * 32 + g * 8;
    v8h lo = *(const v8h*)base;           // K = kk*32 + g*8 + 0..7
    v8h hi = *(const v8h*)(base + 16);    // K = kk*32 + 16 + g*8 + 0..7
    v16h r;
#pragma unroll
    for (int i = 0; i < 8; ++i) { r[i] = lo[i]; r[i + 8] = hi[i]; }
    return r;
}

__launch_bounds__(128)
__global__ void reservoir_kernel(const float* __restrict__ x,
                                 const _Float16* __restrict__ wih0,
                                 const _Float16* __restrict__ whh0,
                                 const _Float16* __restrict__ wih1,
                                 const _Float16* __restrict__ whh1,
                                 const float* __restrict__ bih0,
                                 const float* __restrict__ bih1,
                                 float* __restrict__ out) {
    // per-wave private f16 state of h (row-major [16][HH]); doubles as WMMA-A staging
    __shared__ _Float16 hbuf0[4][16 * HH];
    __shared__ _Float16 hbuf1[4][16 * HH];

    const int wave = threadIdx.x >> 5;
    const int lane = threadIdx.x & 31;
    const int g    = lane >> 4;     // lane group
    const int n16  = lane & 15;     // N index within 16-tile / A row
    const int tile = blockIdx.x * 4 + wave;
    const int m0   = tile * 16;

    // wave-uniform "this wave produces output" flag, as a scalar (SGPR) condition
    const bool activeS = (__builtin_amdgcn_readfirstlane(tile) < (BN / 16));

    // zero the h state (h starts at 0)
    for (int i = lane; i < 16 * HH; i += 32) {
        hbuf0[wave][i] = (_Float16)0.f;
        hbuf1[wave][i] = (_Float16)0.f;
    }
    __builtin_amdgcn_wave_barrier();

    // per-lane x row base: A-matrix row m = n16, bn = m0 + n16 (clamped for tail waves)
    int bnA = m0 + n16; if (bnA > BN - 1) bnA = BN - 1;
    const int bA = bnA / NN, nA = bnA % NN;
    const long xrow = (long)bA * SS * NN * FF + (long)nA * FF;

    // biases: C/D layout has N = n16 in every accumulator VGPR
    float bv0[NTILES], bv1[NTILES];
#pragma unroll
    for (int t = 0; t < NTILES; ++t) {
        bv0[t] = bih0[t * 16 + n16];
        bv1[t] = bih1[t * 16 + n16];
    }

    // 32-bit output row offsets for C/D layout: M = j + 8*g, N-col = t*16 + n16
    int orow[8];
#pragma unroll
    for (int j = 0; j < 8; ++j) {
        int bn = m0 + j + 8 * g;
        int bnc = (bn < BN) ? bn : 0;
        int b = bnc / NN, n = bnc % NN;
        orow[j] = (b * SS * NN + n) * 256;   // + s*NN*256 per step (max ~41M, fits i32)
    }

#pragma unroll 1
    for (int s = 0; s < SS; ++s) {
        // keep the (loop-invariant) weight-fragment loads inside the loop:
        // without this, LICM hoists ~800 VGPRs of B fragments and spills to scratch.
        asm volatile("" ::: "memory");

        // ---- A fragments of h0_old (snapshot before overwrite) ----
        v16h aH0[4];
#pragma unroll
        for (int kk = 0; kk < 4; ++kk) aH0[kk] = load_a_frag(hbuf0[wave], n16, g, kk);

        // ---- A fragment of x_s (global f32 -> f16) ----
        v16h ax;
        {
            const float* xp = x + xrow + (long)s * NN * FF;
            v4f f0 = *(const v4f*)(xp + g * 8);
            v4f f1 = *(const v4f*)(xp + g * 8 + 4);
            v4f f2 = *(const v4f*)(xp + 16 + g * 8);
            v4f f3 = *(const v4f*)(xp + 16 + g * 8 + 4);
#pragma unroll
            for (int i = 0; i < 4; ++i) {
                ax[i]      = (_Float16)f0[i];
                ax[4 + i]  = (_Float16)f1[i];
                ax[8 + i]  = (_Float16)f2[i];
                ax[12 + i] = (_Float16)f3[i];
            }
        }

        const int obase = s * NN * 256;

        // ================= layer 0 =================
#pragma unroll
        for (int t = 0; t < NTILES; ++t) {
            v8f c;
#pragma unroll
            for (int j = 0; j < 8; ++j) c[j] = bv0[t];

            // x @ w_ih0^T : single K=32 block
            v16h bf = load_frag16(wih0 + (t * 16 + n16) * FF + g * 8);
            c = __builtin_amdgcn_wmma_f32_16x16x32_f16(false, ax, false, bf, (short)0, c, false, false);
            // h0_old @ w_hh0^T : 4 K-blocks
#pragma unroll
            for (int kk = 0; kk < 4; ++kk) {
                v16h bh = load_frag16(whh0 + (t * 16 + n16) * HH + kk * 32 + g * 8);
                c = __builtin_amdgcn_wmma_f32_16x16x32_f16(false, aH0[kk], false, bh, (short)0, c, false, false);
            }

            // leaky update: batch the 8 h_old loads, then compute, then store
            float hold[8], hn[8];
#pragma unroll
            for (int j = 0; j < 8; ++j)
                hold[j] = (float)hbuf0[wave][(j + 8 * g) * HH + t * 16 + n16];
#pragma unroll
            for (int j = 0; j < 8; ++j)
                hn[j] = (1.f - ALPHA) * hold[j] + ALPHA * fast_tanh(c[j]);
#pragma unroll
            for (int j = 0; j < 8; ++j)
                hbuf0[wave][(j + 8 * g) * HH + t * 16 + n16] = (_Float16)hn[j];
            if (activeS) {
#pragma unroll
                for (int j = 0; j < 8; ++j)
                    out[orow[j] + obase + t * 16 + n16] = hn[j];
            }
        }
        __builtin_amdgcn_wave_barrier();

        // ---- A fragments: h0_new and h1_old ----
        v16h aH0n[4], aH1[4];
#pragma unroll
        for (int kk = 0; kk < 4; ++kk) {
            aH0n[kk] = load_a_frag(hbuf0[wave], n16, g, kk);
            aH1[kk]  = load_a_frag(hbuf1[wave], n16, g, kk);
        }

        // ================= layer 1 =================
#pragma unroll
        for (int t = 0; t < NTILES; ++t) {
            v8f c;
#pragma unroll
            for (int j = 0; j < 8; ++j) c[j] = bv1[t];
#pragma unroll
            for (int kk = 0; kk < 4; ++kk) {
                v16h bi = load_frag16(wih1 + (t * 16 + n16) * HH + kk * 32 + g * 8);
                c = __builtin_amdgcn_wmma_f32_16x16x32_f16(false, aH0n[kk], false, bi, (short)0, c, false, false);
            }
#pragma unroll
            for (int kk = 0; kk < 4; ++kk) {
                v16h bh = load_frag16(whh1 + (t * 16 + n16) * HH + kk * 32 + g * 8);
                c = __builtin_amdgcn_wmma_f32_16x16x32_f16(false, aH1[kk], false, bh, (short)0, c, false, false);
            }

            float hold[8], hn[8];
#pragma unroll
            for (int j = 0; j < 8; ++j)
                hold[j] = (float)hbuf1[wave][(j + 8 * g) * HH + t * 16 + n16];
#pragma unroll
            for (int j = 0; j < 8; ++j)
                hn[j] = (1.f - ALPHA) * hold[j] + ALPHA * fast_tanh(c[j]);
#pragma unroll
            for (int j = 0; j < 8; ++j)
                hbuf1[wave][(j + 8 * g) * HH + t * 16 + n16] = (_Float16)hn[j];
            if (activeS) {
#pragma unroll
                for (int j = 0; j < 8; ++j)
                    out[orow[j] + obase + 128 + t * 16 + n16] = hn[j];
            }
        }
        __builtin_amdgcn_wave_barrier();
    }
}

extern "C" void kernel_launch(void* const* d_in, const int* in_sizes, int n_in,
                              void* d_out, int out_size, void* d_ws, size_t ws_size,
                              hipStream_t stream) {
    const float* x    = (const float*)d_in[0];
    const float* wih0 = (const float*)d_in[1];
    const float* whh0 = (const float*)d_in[2];
    const float* bih0 = (const float*)d_in[3];
    const float* wih1 = (const float*)d_in[4];
    const float* whh1 = (const float*)d_in[5];
    const float* bih1 = (const float*)d_in[6];
    _Float16* wsH = (_Float16*)d_ws;

    // convert weights to f16 once per launch (tiny; stays hot in L2 for all 200 steps)
    cvt_f32_f16_kernel<<<16, 256, 0, stream>>>(wih0, wsH + OFF_IH0, 128 * 32);
    cvt_f32_f16_kernel<<<64, 256, 0, stream>>>(whh0, wsH + OFF_HH0, 128 * 128);
    cvt_f32_f16_kernel<<<64, 256, 0, stream>>>(wih1, wsH + OFF_IH1, 128 * 128);
    cvt_f32_f16_kernel<<<64, 256, 0, stream>>>(whh1, wsH + OFF_HH1, 128 * 128);

    // 50 row-tiles of 16 rows, 4 waves per block -> 13 blocks, persistent over all 200 steps
    reservoir_kernel<<<13, 128, 0, stream>>>(x,
                                             wsH + OFF_IH0, wsH + OFF_HH0,
                                             wsH + OFF_IH1, wsH + OFF_HH1,
                                             bih0, bih1, (float*)d_out);
}
